// CrossMambaBlock_33311766347948
// MI455X (gfx1250) — matrix operand order repaired and checked by reading, not compiled, not run on hardware
//
#include <hip/hip_runtime.h>

// CrossMambaBlock for MI455X (gfx1250, wave32, WMMA + async-to-LDS).
#define BB   4
#define LL   16384
#define DD   64
#define NNs  16
#define BLr  (BB * LL)        // 65536 rows
#define NCk  64               // scan chunks
#define CSz  (LL / NCk)       // 256 steps per chunk
#define TIL  16               // scan staging tile (timesteps)
#define PKF  176              // floats per staged timestep: dt(64)+u(64)+dbl(48)

typedef __attribute__((ext_vector_type(16))) _Float16 v16h;
typedef __attribute__((ext_vector_type(8)))  float    v8f;
typedef __attribute__((ext_vector_type(2)))  float    v2f;

// ---------- scalar helpers ----------
__device__ __forceinline__ float ntn(float x, float nv, float pv, float nf) {
  if (x != x) return nv;
  if (x == __builtin_inff()) return pv;
  if (x == -__builtin_inff()) return nf;
  return x;
}
__device__ __forceinline__ float sigm(float x) { return 1.f / (1.f + __expf(-x)); }
__device__ __forceinline__ float silu_f(float x) { return x * sigm(x); }
__device__ __forceinline__ float softplus_f(float x) { return x > 20.f ? x : log1pf(__expf(x)); }

// ---------- async-to-LDS helpers (CDNA5) ----------
__device__ __forceinline__ void async_b128(unsigned laddr, const void* gptr) {
  asm volatile("global_load_async_to_lds_b128 %0, %1, off"
               :: "v"(laddr), "v"((unsigned long long)(uintptr_t)gptr) : "memory");
}
__device__ __forceinline__ void wait_async0() {
  asm volatile("s_wait_asynccnt 0x0" ::: "memory");
}
// Wave 0 stages TIL timesteps of {dt row, u row, dbl row} into LDS at slotBase.
// Per timestep pack: dt @ +0 (256B), u @ +256, dbl @ +512 (192B); stride 704B.
__device__ __forceinline__ void stage_tile(const float* dt, const float* u, const float* dbl,
                                           size_t rowT, unsigned slotBase, int lane) {
#pragma unroll
  for (int i = 0; i < 8; ++i) {                 // dt + u: 16 chunks/row, 256 chunks each
    int c = i * 32 + lane;
    int toff = c >> 4, cir = c & 15;
    async_b128(slotBase + toff * 704 + cir * 16,       dt + (rowT + toff) * DD + cir * 4);
    async_b128(slotBase + toff * 704 + 256 + cir * 16, u  + (rowT + toff) * DD + cir * 4);
  }
#pragma unroll
  for (int i = 0; i < 6; ++i) {                 // dbl: 12 chunks/row, 192 chunks
    int c = i * 32 + lane;
    int toff = c / 12, cir = c % 12;
    async_b128(slotBase + toff * 704 + 512 + cir * 16, dbl + (rowT + toff) * 48 + cir * 4);
  }
}

// ---------- WMMA helpers (wave32) ----------
__device__ __forceinline__ v8f wmma16(v16h a, v16h b, v8f c) {
  return __builtin_amdgcn_wmma_f32_16x16x32_f16(false, a, false, b, (short)0, c, false, false);
}
__device__ __forceinline__ v8f wmma4f32(v2f a, v2f b, v8f c) {
  return __builtin_amdgcn_wmma_f32_16x16x4_f32(false, a, false, b, (short)0, c, false, false);
}
// A: 16 rows x 32 K from row-major f32 [.., 64]; lane=row; contiguous K layout (k = 16*hi + s).
__device__ __forceinline__ v16h load_a_frag(const float* src, int rowBase, int kBase, int lane) {
  int r = lane & 15, hi = (lane >> 4) & 1;
  const float* p = src + (size_t)(rowBase + r) * DD + kBase;
  v16h a;
#pragma unroll
  for (int h = 0; h < 8; ++h) a[h] = (_Float16)p[hi * 16 + h];
#pragma unroll
  for (int h = 0; h < 8; ++h) a[8 + h] = (_Float16)p[hi * 16 + 8 + h];
  return a;
}
// Stage weight W[Ndim][Kdim] into LDS as packed B-fragments (32B contiguous per lane,
// so a lane fetches its whole v16h with ds_load_b128 x2).
__device__ __forceinline__ void stage_b_packed(_Float16* dst, const float* __restrict__ W,
                                               int Kdim, int Ndim, int NT, int nFrag, int tid) {
  for (int idx = tid; idx < nFrag * 512; idx += 256) {
    int f = idx >> 9, r = idx & 511;
    int lane = r >> 4, h = r & 15;
    int kt = f / NT, nt = f % NT;
    int n = nt * 16 + (lane & 15);
    int k = kt * 32 + ((lane >> 4) << 4) + h;
    float v = (k < Kdim && n < Ndim) ? W[n * Kdim + k] : 0.f;
    dst[idx] = (_Float16)v;
  }
}
__device__ __forceinline__ v16h load_b_packed(const _Float16* Bsh, int f, int lane) {
  return *reinterpret_cast<const v16h*>(Bsh + f * 512 + lane * 16);
}
// C/D: f32 16x16; VGPR v -> row v + hi*8, lane&15 -> col.
__device__ __forceinline__ void store_c(float* dst, int rowBase, int colBase, int ldo, v8f c, int lane) {
  int n = lane & 15, hi = (lane >> 4) & 1;
#pragma unroll
  for (int v = 0; v < 8; ++v)
    dst[(size_t)(rowBase + v + hi * 8) * ldo + colBase + n] = c[v];
}

// ---------- 1. input layernorms ----------
__global__ void k_ln_inputs(const float* __restrict__ in0, const float* __restrict__ in1,
                            const float* __restrict__ w0, const float* __restrict__ b0,
                            const float* __restrict__ w1, const float* __restrict__ b1,
                            float* __restrict__ x0, float* __restrict__ x1) {
  size_t row = (size_t)blockIdx.x * blockDim.x + threadIdx.x;
  if (row >= (size_t)BLr) return;
  const float* s0 = in0 + row * DD;
  const float* s1 = in1 + row * DD;
  float m = 0.f, q = 0.f;
  for (int i = 0; i < DD; ++i) { float v = ntn(s0[i], 0.f, 1.f, -1.f); m += v; q += v * v; }
  m *= (1.f / DD); float rs = rsqrtf(q * (1.f / DD) - m * m + 1e-5f);
  for (int i = 0; i < DD; ++i) {
    float v = ntn(s0[i], 0.f, 1.f, -1.f);
    x0[row * DD + i] = ntn((v - m) * rs * w0[i] + b0[i], 0.f, 1.f, -1.f);
  }
  m = 0.f; q = 0.f;
  for (int i = 0; i < DD; ++i) { float v = ntn(s1[i], 0.f, 1.f, -1.f); m += v; q += v * v; }
  m *= (1.f / DD); rs = rsqrtf(q * (1.f / DD) - m * m + 1e-5f);
  for (int i = 0; i < DD; ++i) {
    float v = ntn(s1[i], 0.f, 1.f, -1.f);
    x1[row * DD + i] = ntn((v - m) * rs * w1[i] + b1[i], 0.f, 1.f, -1.f);
  }
}

// ---------- 2. fused projection GEMMs ----------
__global__ __launch_bounds__(256) void k_proj(const float* __restrict__ x0, const float* __restrict__ x1,
                                              const float* __restrict__ cw_w,
                                              const float* __restrict__ ip_w,
                                              const float* __restrict__ ipe_w,
                                              float* wpre, float* xf, float* zf, float* xe, float* ze) {
  __shared__ __align__(32) _Float16 Bsh[48 * 512];
  _Float16* Bcw  = Bsh;
  _Float16* Bip  = Bsh + 16 * 512;
  _Float16* Bipe = Bsh + 32 * 512;
  int tid = threadIdx.x;
  stage_b_packed(Bcw,  cw_w, 128,  64, 4, 16, tid);
  stage_b_packed(Bip,  ip_w,  64, 128, 8, 16, tid);
  stage_b_packed(Bipe, ipe_w, 64, 128, 8, 16, tid);
  __syncthreads();
  int lane = tid & 31, wave = tid >> 5;
  int rowBase = blockIdx.x * 128 + wave * 16;
  v16h a0lo = load_a_frag(x0, rowBase, 0, lane);
  v16h a0hi = load_a_frag(x0, rowBase, 32, lane);
  v16h a1lo = load_a_frag(x1, rowBase, 0, lane);
  v16h a1hi = load_a_frag(x1, rowBase, 32, lane);
#pragma unroll
  for (int nt = 0; nt < 4; ++nt) {      // cross-gate pre-activation (K=128 over [x0|x1])
    v8f acc = {};
    acc = wmma16(a0lo, load_b_packed(Bcw, 0 * 4 + nt, lane), acc);
    acc = wmma16(a0hi, load_b_packed(Bcw, 1 * 4 + nt, lane), acc);
    acc = wmma16(a1lo, load_b_packed(Bcw, 2 * 4 + nt, lane), acc);
    acc = wmma16(a1hi, load_b_packed(Bcw, 3 * 4 + nt, lane), acc);
    store_c(wpre, rowBase, nt * 16, 64, acc, lane);
  }
#pragma unroll
  for (int nt = 0; nt < 8; ++nt) {      // in_proj: cols 0..63 -> x, 64..127 -> z
    v8f acc = {};
    acc = wmma16(a0lo, load_b_packed(Bip, 0 * 8 + nt, lane), acc);
    acc = wmma16(a0hi, load_b_packed(Bip, 1 * 8 + nt, lane), acc);
    store_c(nt < 4 ? xf : zf, rowBase, (nt & 3) * 16, 64, acc, lane);
  }
#pragma unroll
  for (int nt = 0; nt < 8; ++nt) {      // in_proj_e
    v8f acc = {};
    acc = wmma16(a1lo, load_b_packed(Bipe, 0 * 8 + nt, lane), acc);
    acc = wmma16(a1hi, load_b_packed(Bipe, 1 * 8 + nt, lane), acc);
    store_c(nt < 4 ? xe : ze, rowBase, (nt & 3) * 16, 64, acc, lane);
  }
}

// ---------- 3. gate finalize: LN + sigmoid + clamp ----------
__global__ void k_gate(const float* wpre, const float* __restrict__ lin_b,
                       const float* __restrict__ nw, const float* __restrict__ nb, float* wout) {
  size_t row = (size_t)blockIdx.x * blockDim.x + threadIdx.x;
  if (row >= (size_t)BLr) return;
  float m = 0.f, q = 0.f;
  for (int i = 0; i < DD; ++i) { float v = wpre[row * DD + i] + lin_b[i]; m += v; q += v * v; }
  m *= (1.f / DD); float rs = rsqrtf(q * (1.f / DD) - m * m + 1e-5f);
  for (int i = 0; i < DD; ++i) {
    float v = wpre[row * DD + i] + lin_b[i];
    float g = ntn((v - m) * rs * nw[i] + nb[i], 0.5f, 1.f, 0.f);
    g = sigm(g);
    wout[row * DD + i] = fminf(fmaxf(g, 0.01f), 0.99f);
  }
}

// ---------- 4. depthwise causal conv + SiLU (flip=1 -> flipped time) ----------
__global__ void k_conv(const float* __restrict__ xin, const float* __restrict__ cw,
                       const float* __restrict__ cb, float* __restrict__ uout, int flip) {
  size_t g = (size_t)blockIdx.x * blockDim.x + threadIdx.x;
  if (g >= (size_t)BLr * DD) return;
  int d = (int)(g & 63);
  size_t bt = g >> 6;
  int b = (int)(bt / LL), t = (int)(bt % LL);
  float acc = cb[d];
#pragma unroll
  for (int k = 0; k < 4; ++k) {
    int tt = t - 3 + k;
    if (tt >= 0) {
      int st = flip ? (LL - 1 - tt) : tt;
      acc += cw[d * 4 + k] * xin[((size_t)b * LL + st) * DD + d];
    }
  }
  uout[g] = silu_f(acc);
}

// ---------- 5. x-projection GEMM: dbl = u @ xproj^T, 36 cols padded to 48 ----------
__global__ __launch_bounds__(256) void k_xproj(const float* __restrict__ u,
                                               const float* __restrict__ xp_w,
                                               float* dbl) {
  __shared__ __align__(32) _Float16 Bxp[6 * 512];   // KT=2, NT=3
  int tid = threadIdx.x;
  stage_b_packed(Bxp, xp_w, 64, 36, 3, 6, tid);
  __syncthreads();
  int lane = tid & 31, wave = tid >> 5;
  int rowBase = blockIdx.x * 128 + wave * 16;
  v16h alo = load_a_frag(u, rowBase, 0, lane);
  v16h ahi = load_a_frag(u, rowBase, 32, lane);
#pragma unroll
  for (int nt = 0; nt < 3; ++nt) {
    v8f acc = {};
    acc = wmma16(alo, load_b_packed(Bxp, 0 * 3 + nt, lane), acc);
    acc = wmma16(ahi, load_b_packed(Bxp, 1 * 3 + nt, lane), acc);
    store_c(dbl, rowBase, nt * 16, 48, acc, lane);
  }
}

// ---------- 6. dt = softplus(dt_lowrank @ dtproj^T + b) via V_WMMA_F32_16X16X4_F32 ----------
__global__ __launch_bounds__(256) void k_dt(const float* __restrict__ dbl,
                                            const float* __restrict__ dtw,
                                            const float* __restrict__ dtb,
                                            float* __restrict__ dtout) {
  __shared__ __align__(8) float Bdt[4 * 64];
  __shared__ float biasSh[64];
  int tid = threadIdx.x;
  if (tid < 64) biasSh[tid] = dtb[tid];
  if (tid < 256) {
    int f = tid >> 6, r = tid & 63;
    int lane = r >> 1, h = r & 1;
    int n = f * 16 + (lane & 15);
    int k = ((lane >> 4) << 1) + h;
    Bdt[tid] = dtw[n * 4 + k];
  }
  __syncthreads();
  int lane = tid & 31, wave = tid >> 5;
  int rowBase = blockIdx.x * 128 + wave * 16;
  int r = lane & 15, hi = (lane >> 4) & 1;
  v2f a;
  a[0] = dbl[(size_t)(rowBase + r) * 48 + hi * 2 + 0];
  a[1] = dbl[(size_t)(rowBase + r) * 48 + hi * 2 + 1];
#pragma unroll
  for (int nt = 0; nt < 4; ++nt) {
    v2f bfr = *reinterpret_cast<const v2f*>(Bdt + nt * 64 + lane * 2);
    v8f acc = {};
    acc = wmma4f32(a, bfr, acc);
    int col = nt * 16 + (lane & 15);
    float bias = biasSh[col];
#pragma unroll
    for (int v = 0; v < 8; ++v)
      dtout[(size_t)(rowBase + v + hi * 8) * DD + col] = softplus_f(acc[v] + bias);
  }
}

// ---------- 7-9. chunked selective scan with async-DMA operand staging ----------
__global__ __launch_bounds__(1024) void k_scan1(const float* __restrict__ dt,
                                                const float* __restrict__ u,
                                                const float* __restrict__ dbl,
                                                const float* __restrict__ A_log,
                                                float* __restrict__ aProd, float* __restrict__ hEnd) {
  __shared__ __align__(16) float ring[2 * TIL * PKF];
  int tid = threadIdx.x;
  int d = tid >> 4, n = tid & 15, lane = tid & 31;
  int c = blockIdx.x, b = blockIdx.y;
  float Ad = -__expf(A_log[d * NNs + n]);
  float h = 0.f, slog = 0.f;
  size_t row0 = (size_t)b * LL + (size_t)c * CSz;
  unsigned lds0 = (unsigned)(uintptr_t)ring;
  const unsigned BUFB = TIL * PKF * 4;
  if (tid < 32) { stage_tile(dt, u, dbl, row0, lds0, lane); wait_async0(); }
  __syncthreads();
  for (int tt = 0; tt < CSz; tt += TIL) {
    int buf = (tt / TIL) & 1;
    if (tid < 32 && tt + TIL < CSz)
      stage_tile(dt, u, dbl, row0 + tt + TIL, lds0 + (buf ^ 1) * BUFB, lane);
    const float* pk0 = ring + buf * (TIL * PKF);
#pragma unroll
    for (int j = 0; j < TIL; ++j) {
      const float* pk = pk0 + j * PKF;
      float dtv = pk[d];
      float uv  = pk[64 + d];
      float Bv  = pk[128 + 4 + n];
      float da = dtv * Ad;
      slog += da;
      h = __expf(da) * h + dtv * Bv * uv;
    }
    if (tid < 32 && tt + TIL < CSz) wait_async0();
    __syncthreads();
  }
  size_t idx = (((size_t)b * NCk) + c) * 1024 + tid;
  aProd[idx] = __expf(slog);
  hEnd[idx] = h;
}

__global__ __launch_bounds__(1024) void k_scan2(const float* __restrict__ aProd,
                                                const float* __restrict__ hEnd,
                                                float* __restrict__ carryIn) {
  int tid = threadIdx.x;
  int b = blockIdx.x;
  float carry = 0.f;
  for (int c = 0; c < NCk; ++c) {
    size_t idx = (((size_t)b * NCk) + c) * 1024 + tid;
    carryIn[idx] = carry;
    carry = aProd[idx] * carry + hEnd[idx];
  }
}

__global__ __launch_bounds__(1024) void k_scan3(const float* __restrict__ dt,
                                                const float* __restrict__ u,
                                                const float* __restrict__ dbl,
                                                const float* __restrict__ A_log,
                                                const float* __restrict__ Dvec,
                                                const float* __restrict__ carryIn,
                                                float* __restrict__ y) {
  __shared__ __align__(16) float ring[2 * TIL * PKF];
  int tid = threadIdx.x;
  int d = tid >> 4, n = tid & 15, lane = tid & 31;
  int c = blockIdx.x, b = blockIdx.y;
  float Ad = -__expf(A_log[d * NNs + n]);
  float h = carryIn[(((size_t)b * NCk) + c) * 1024 + tid];
  float Dd = Dvec[d];
  size_t row0 = (size_t)b * LL + (size_t)c * CSz;
  unsigned lds0 = (unsigned)(uintptr_t)ring;
  const unsigned BUFB = TIL * PKF * 4;
  if (tid < 32) { stage_tile(dt, u, dbl, row0, lds0, lane); wait_async0(); }
  __syncthreads();
  for (int tt = 0; tt < CSz; tt += TIL) {
    int buf = (tt / TIL) & 1;
    if (tid < 32 && tt + TIL < CSz)
      stage_tile(dt, u, dbl, row0 + tt + TIL, lds0 + (buf ^ 1) * BUFB, lane);
    const float* pk0 = ring + buf * (TIL * PKF);
#pragma unroll
    for (int j = 0; j < TIL; ++j) {
      const float* pk = pk0 + j * PKF;
      float dtv = pk[d];
      float uv  = pk[64 + d];
      float Bv  = pk[128 + 4 + n];
      float Cv  = pk[128 + 20 + n];
      h = __expf(dtv * Ad) * h + dtv * Bv * uv;
      float p = h * Cv;                  // reduce over n: 16-lane half-wave (wave32)
      p += __shfl_xor(p, 1, 32);
      p += __shfl_xor(p, 2, 32);
      p += __shfl_xor(p, 4, 32);
      p += __shfl_xor(p, 8, 32);
      if (n == 0) y[(row0 + tt + j) * DD + d] = p + uv * Dd;
    }
    if (tid < 32 && tt + TIL < CSz) wait_async0();
    __syncthreads();
  }
}

// ---------- 10. combine branches + RMS norm ----------
__global__ void k_combine(const float* yf, const float* __restrict__ yr,
                          const float* __restrict__ z, const float* __restrict__ ze,
                          const float* __restrict__ mnw, float* g) {
  size_t row = (size_t)blockIdx.x * blockDim.x + threadIdx.x;
  if (row >= (size_t)BLr) return;
  size_t b = row / LL, t = row % LL;
  size_t mrow = b * LL + (LL - 1 - t);   // un-flip reverse branch
  float q = 0.f;
  for (int i = 0; i < DD; ++i) {
    float v = (yf[row * DD + i] * silu_f(z[row * DD + i]) +
               yr[mrow * DD + i] * silu_f(ze[row * DD + i])) * 0.5f;
    q += v * v;
  }
  float rs = rsqrtf(q * (1.f / DD) + 1e-5f);
  for (int i = 0; i < DD; ++i) {
    float v = (yf[row * DD + i] * silu_f(z[row * DD + i]) +
               yr[mrow * DD + i] * silu_f(ze[row * DD + i])) * 0.5f;
    g[row * DD + i] = v * rs * mnw[i];
  }
}

// ---------- 11. out_proj GEMM ----------
__global__ __launch_bounds__(256) void k_outproj(const float* __restrict__ g,
                                                 const float* __restrict__ op_w,
                                                 float* o) {
  __shared__ __align__(32) _Float16 Bop[8 * 512];   // KT=2, NT=4
  int tid = threadIdx.x;
  stage_b_packed(Bop, op_w, 64, 64, 4, 8, tid);
  __syncthreads();
  int lane = tid & 31, wave = tid >> 5;
  int rowBase = blockIdx.x * 128 + wave * 16;
  v16h alo = load_a_frag(g, rowBase, 0, lane);
  v16h ahi = load_a_frag(g, rowBase, 32, lane);
#pragma unroll
  for (int nt = 0; nt < 4; ++nt) {
    v8f acc = {};
    acc = wmma16(alo, load_b_packed(Bop, 0 * 4 + nt, lane), acc);
    acc = wmma16(ahi, load_b_packed(Bop, 1 * 4 + nt, lane), acc);
    store_c(o, rowBase, nt * 16, 64, acc, lane);
  }
}

// ---------- 12. post LN + gate + residual ----------
__global__ void k_final(const float* __restrict__ o, const float* __restrict__ wg,
                        const float* __restrict__ x0, const float* __restrict__ in0,
                        const float* __restrict__ pw, const float* __restrict__ pb,
                        float* __restrict__ out) {
  size_t row = (size_t)blockIdx.x * blockDim.x + threadIdx.x;
  if (row >= (size_t)BLr) return;
  float m = 0.f, q = 0.f;
  for (int i = 0; i < DD; ++i) { float v = o[row * DD + i]; m += v; q += v * v; }
  m *= (1.f / DD); float rs = rsqrtf(q * (1.f / DD) - m * m + 1e-5f);
  for (int i = 0; i < DD; ++i) {
    float v = ntn((o[row * DD + i] - m) * rs * pw[i] + pb[i], 0.f, 1.f, -1.f);
    float wv = wg[row * DD + i];
    float skip = ntn(in0[row * DD + i], 0.f, 1.f, -1.f);
    out[row * DD + i] = v * wv + x0[row * DD + i] * (1.f - wv) + skip;
  }
}

extern "C" void kernel_launch(void* const* d_in, const int* in_sizes, int n_in,
                              void* d_out, int out_size, void* d_ws, size_t ws_size,
                              hipStream_t stream) {
  (void)in_sizes; (void)n_in; (void)out_size; (void)ws_size;
  const float* input0     = (const float*)d_in[0];
  const float* input1     = (const float*)d_in[1];
  const float* norm0_w    = (const float*)d_in[2];
  const float* norm0_b    = (const float*)d_in[3];
  const float* norm1_w    = (const float*)d_in[4];
  const float* norm1_b    = (const float*)d_in[5];
  const float* cw_lin_w   = (const float*)d_in[6];
  const float* cw_lin_b   = (const float*)d_in[7];
  const float* cw_norm_w  = (const float*)d_in[8];
  const float* cw_norm_b  = (const float*)d_in[9];
  const float* in_proj_w  = (const float*)d_in[10];
  const float* in_proj_e_w= (const float*)d_in[11];
  const float* conv_w_f   = (const float*)d_in[12];
  const float* conv_b_f   = (const float*)d_in[13];
  const float* xproj_w_f  = (const float*)d_in[14];
  const float* dtproj_w_f = (const float*)d_in[15];
  const float* dtproj_b_f = (const float*)d_in[16];
  const float* A_log_f    = (const float*)d_in[17];
  const float* D_f        = (const float*)d_in[18];
  const float* conv_w_r   = (const float*)d_in[19];
  const float* conv_b_r   = (const float*)d_in[20];
  const float* xproj_w_r  = (const float*)d_in[21];
  const float* dtproj_w_r = (const float*)d_in[22];
  const float* dtproj_b_r = (const float*)d_in[23];
  const float* A_log_r    = (const float*)d_in[24];
  const float* D_r        = (const float*)d_in[25];
  const float* mamba_nw   = (const float*)d_in[26];
  const float* out_proj_w = (const float*)d_in[27];
  const float* post_nw    = (const float*)d_in[28];
  const float* post_nb    = (const float*)d_in[29];

  // workspace layout (floats); planes of BL*64 with lifetime-based reuse
  float* W = (float*)d_ws;
  const size_t P = (size_t)BLr * DD;
  float* x0   = W + 0 * P;
  float* x1   = W + 1 * P;   // reused as u_f after k_proj consumes x1
  float* wg   = W + 2 * P;
  float* xbuf = W + 3 * P;   // x branch input; reused as dbl_f (stride 48 < 64)
  float* zbuf = W + 4 * P;
  float* xe   = W + 5 * P;   // x_e branch input; reused as dbl_r
  float* ze   = W + 6 * P;
  float* ur   = W + 7 * P;
  float* dtf  = W + 8 * P;
  float* dtr  = W + 9 * P;
  float* yf   = W + 10 * P;  // reused as g (rms-normed)
  float* yr   = W + 11 * P;  // reused as o (out_proj result)
  const size_t CSZ = (size_t)BB * NCk * 1024;   // per-direction chunk-state lanes
  float* aP = W + 12 * P;           // [2][CSZ]
  float* hE = aP + 2 * CSZ;         // [2][CSZ]
  float* cI = aP + 4 * CSZ;         // [2][CSZ]

  const dim3 blk256(256);
  const int rowsGrid = (BLr + 255) / 256;
  const int gemmGrid = BLr / 128;
  const size_t elemN = (size_t)BLr * DD;
  const int elemGrid = (int)((elemN + 255) / 256);

  k_ln_inputs<<<rowsGrid, blk256, 0, stream>>>(input0, input1, norm0_w, norm0_b,
                                               norm1_w, norm1_b, x0, x1);
  k_proj<<<gemmGrid, blk256, 0, stream>>>(x0, x1, cw_lin_w, in_proj_w, in_proj_e_w,
                                          wg, xbuf, zbuf, xe, ze);
  k_gate<<<rowsGrid, blk256, 0, stream>>>(wg, cw_lin_b, cw_norm_w, cw_norm_b, wg);
  float* uf = x1;
  k_conv<<<elemGrid, blk256, 0, stream>>>(xbuf, conv_w_f, conv_b_f, uf, 0);
  k_conv<<<elemGrid, blk256, 0, stream>>>(xe,   conv_w_r, conv_b_r, ur, 1);
  float* dblf = xbuf;
  float* dblr = xe;
  k_xproj<<<gemmGrid, blk256, 0, stream>>>(uf, xproj_w_f, dblf);
  k_xproj<<<gemmGrid, blk256, 0, stream>>>(ur, xproj_w_r, dblr);
  k_dt<<<gemmGrid, blk256, 0, stream>>>(dblf, dtproj_w_f, dtproj_b_f, dtf);
  k_dt<<<gemmGrid, blk256, 0, stream>>>(dblr, dtproj_w_r, dtproj_b_r, dtr);
  dim3 sg(NCk, BB);
  dim3 sb(1024);
  k_scan1<<<sg, sb, 0, stream>>>(dtf, uf, dblf, A_log_f, aP,       hE);
  k_scan1<<<sg, sb, 0, stream>>>(dtr, ur, dblr, A_log_r, aP + CSZ, hE + CSZ);
  k_scan2<<<dim3(BB), sb, 0, stream>>>(aP,       hE,       cI);
  k_scan2<<<dim3(BB), sb, 0, stream>>>(aP + CSZ, hE + CSZ, cI + CSZ);
  k_scan3<<<sg, sb, 0, stream>>>(dtf, uf, dblf, A_log_f, D_f, cI,       yf);
  k_scan3<<<sg, sb, 0, stream>>>(dtr, ur, dblr, A_log_r, D_r, cI + CSZ, yr);
  float* g = yf;
  k_combine<<<rowsGrid, blk256, 0, stream>>>(yf, yr, zbuf, ze, mamba_nw, g);
  float* o = yr;
  k_outproj<<<gemmGrid, blk256, 0, stream>>>(g, out_proj_w, o);
  k_final<<<rowsGrid, blk256, 0, stream>>>(o, wg, x0, input0, post_nw, post_nb, (float*)d_out);
}
